// attention_9887014715880
// MI455X (gfx1250) — matrix operand order repaired and checked by reading, not compiled, not run on hardware
//
#include <hip/hip_runtime.h>
#include <hip/hip_bf16.h>
#include <stdint.h>

// ---------------------------------------------------------------------------
// Autocorrelation attention for MI455X (gfx1250), wave32 + WMMA.
// B=4, H=8, L=4096, E=64.
// ---------------------------------------------------------------------------

typedef __attribute__((ext_vector_type(16))) __bf16    v16bf;
typedef __attribute__((ext_vector_type(8)))  float     v8f;
typedef __attribute__((ext_vector_type(8)))  uint32_t  v8u;

#define LMASK  4095
#define QPITCH 164                 // u32 words per Q row: 128 slots + 30 mirror + pad; %4==0
#define KPITCH 20                  // u32 words per K row: 16 pairs + pad; %4==0
#define KOFF   (64 * QPITCH)       // u32 offset of K region in smem

// round-to-nearest-even f32 -> bf16, packed pair (lo = a, hi = b)
__device__ __forceinline__ uint32_t pack_bf2(float a, float b) {
  uint32_t ua = __builtin_bit_cast(uint32_t, a);
  uint32_t ub = __builtin_bit_cast(uint32_t, b);
  ua = ua + 0x7fffu + ((ua >> 16) & 1u);
  ub = ub + 0x7fffu + ((ub >> 16) & 1u);
  return (ua >> 16) | (ub & 0xffff0000u);
}

__device__ __forceinline__ v8f wmma_bf16(v8u a, v8u b, v8f c) {
  return __builtin_amdgcn_wmma_f32_16x16x32_bf16(
      false, __builtin_bit_cast(v16bf, a),
      false, __builtin_bit_cast(v16bf, b),
      (short)0, c, false, false);
}

// ---------------------------------------------------------------------------
// Kernel 1: corr[b,h,e,tau] = sum_t K[t,e] * Q[(t+tau)%L, e] via bf16 WMMA,
// diagonal extraction. D = A x B with A = Q-rows (M=e, K=t), B = K (K=t, N=e).
// Block = one (b,h) x 64-tau tile, 512 threads = 16 waves, 4 taus/wave.
// ---------------------------------------------------------------------------
__global__ __launch_bounds__(512, 1) void corr_wmma_kernel(
    const float* __restrict__ Q, const float* __restrict__ K,
    float* __restrict__ corr)
{
  // Q: overlapping bf16-pair pack QT[e][slot] = (Q[r], Q[r+1]), slot = rel&127,
  //    cols 128..157 mirror slots 0..29 so 8-word runs never wrap.
  // K: non-overlapping pairs KT[e][j] = (K[t0+2j], K[t0+2j+1]).
  __shared__ __align__(16) uint32_t smem[64 * QPITCH + 64 * KPITCH]; // 47,104 B

  const int tid    = threadIdx.x;
  const int wid    = tid >> 5;        // wave 0..15
  const int lid    = tid & 31;
  const int e_lane = lid & 15;
  const int gsel   = lid >> 4;        // half-wave group

  const int bh   = blockIdx.x >> 6;        // 0..31
  const int tau0 = (blockIdx.x & 63) << 6; // 64 taus per block
  const int base = bh << 18;               // bh * L * E

  v8f acc[4][4];                            // [e-tile][local tau]
  const v8f vzero = {0.f,0.f,0.f,0.f,0.f,0.f,0.f,0.f};
  #pragma unroll
  for (int a = 0; a < 4; ++a)
    #pragma unroll
    for (int b = 0; b < 4; ++b) acc[a][b] = vzero;

  for (int tc = 0; tc < 128; ++tc) {        // 32-wide t chunks
    const int t0 = tc << 5;

    // stage K chunk
    for (int it = tid; it < 64 * 16; it += 512) {
      const int e = it & 63, j = it >> 6;
      const int r = t0 + (j << 1);          // <= 4094, no wrap
      smem[KOFF + e * KPITCH + j] =
          pack_bf2(K[base + (r << 6) + e], K[base + ((r + 1) << 6) + e]);
    }
    // stage Q rows: rolling 128-slot window (chunk 0: 96 rows, then 32/chunk)
    const int nrows = (tc == 0) ? 96 : 32;
    const int rbase = (tc == 0) ? 0 : (t0 + 64);
    for (int it = tid; it < (nrows << 6); it += 512) {
      const int e   = it & 63;
      const int rel = rbase + (it >> 6);
      const int g0  = (tau0 + rel) & LMASK;
      const int g1  = (tau0 + rel + 1) & LMASK;
      const uint32_t pv = pack_bf2(Q[base + (g0 << 6) + e], Q[base + (g1 << 6) + e]);
      const int slot = rel & 127;
      smem[e * QPITCH + slot] = pv;
      if (slot < 30) smem[e * QPITCH + slot + 128] = pv;  // mirror for wrap-free runs
      if (tc + 1 < 128) {                                 // prefetch next chunk rows
        const int gp = (tau0 + rel + 32) & LMASK;
        __builtin_prefetch(&Q[base + (gp << 6) + e], 0, 1);
      }
    }
    __syncthreads();

    // run base in Q cols: (t0 + 4*wid) & 127 is a multiple of 4; +8*gsel keeps %4==0
    const int sb0 = ((t0 + (wid << 2)) & 127) + (gsel << 3);

    #pragma unroll
    for (int et = 0; et < 4; ++et) {
      const uint32_t* qrow = &smem[(et * 16 + e_lane) * QPITCH];
      const uint32_t* krow = &smem[KOFF + (et * 16 + e_lane) * KPITCH];

      // B from K: words 4g..4g+3 and 8+4g..8+4g+3 -> 2 x ds_load_b128 (16B aligned)
      const uint4 b0 = *(const uint4*)(krow + (gsel << 2));
      const uint4 b1 = *(const uint4*)(krow + (gsel << 2) + 8);
      const v8u bu = {b0.x, b0.y, b0.z, b0.w, b1.x, b1.y, b1.z, b1.w};

      // tau pair j = {0,1}: col base %4==0 -> 4 x ds_load_b128; even/odd words
      // are the A operands of tau j=0 / j=1 respectively.
      {
        const int cb = sb0;
        const uint4 q0 = *(const uint4*)(qrow + cb);
        const uint4 q1 = *(const uint4*)(qrow + cb + 4);
        const uint4 q2 = *(const uint4*)(qrow + cb + 16);
        const uint4 q3 = *(const uint4*)(qrow + cb + 20);
        const v8u ae = {q0.x, q0.z, q1.x, q1.z, q2.x, q2.z, q3.x, q3.z};
        const v8u ao = {q0.y, q0.w, q1.y, q1.w, q2.y, q2.w, q3.y, q3.w};
        acc[et][0] = wmma_bf16(ae, bu, acc[et][0]);
        acc[et][1] = wmma_bf16(ao, bu, acc[et][1]);
      }
      // tau pair j = {2,3}: col base %4==2 -> 8 x ds_load_b64 (8B aligned)
      {
        const int cb = sb0 + 2;
        const uint2 u0 = *(const uint2*)(qrow + cb);
        const uint2 u1 = *(const uint2*)(qrow + cb + 2);
        const uint2 u2 = *(const uint2*)(qrow + cb + 4);
        const uint2 u3 = *(const uint2*)(qrow + cb + 6);
        const uint2 u4 = *(const uint2*)(qrow + cb + 16);
        const uint2 u5 = *(const uint2*)(qrow + cb + 18);
        const uint2 u6 = *(const uint2*)(qrow + cb + 20);
        const uint2 u7 = *(const uint2*)(qrow + cb + 22);
        const v8u ae = {u0.x, u1.x, u2.x, u3.x, u4.x, u5.x, u6.x, u7.x};
        const v8u ao = {u0.y, u1.y, u2.y, u3.y, u4.y, u5.y, u6.y, u7.y};
        acc[et][2] = wmma_bf16(ae, bu, acc[et][2]);
        acc[et][3] = wmma_bf16(ao, bu, acc[et][3]);
      }
    }
    __syncthreads();
  }

  // diagonal extraction: C/D layout -> VGPR v holds (M=v+8*gsel, N=lane%16).
  // Scratch aliases the (dead) Q staging region.
  float* dsc = (float*)smem;   // 16 waves * 16 * 17 floats = 17,408 B
  #pragma unroll
  for (int et = 0; et < 4; ++et) {
    #pragma unroll
    for (int j = 0; j < 4; ++j) {
      #pragma unroll
      for (int v = 0; v < 8; ++v)
        dsc[(wid * 16 + v + (gsel << 3)) * 17 + e_lane] = acc[et][j][v];
      __syncthreads();
      if (lid < 16) {
        const int e   = (et << 4) + lid;
        const int tau = tau0 + (wid << 2) + j;
        corr[(((bh << 6) + e) << 12) + tau] = dsc[(wid * 16 + lid) * 17 + lid];
      }
      __syncthreads();
    }
  }
}

// ---------------------------------------------------------------------------
// Kernel 2: per-row softmax over tau, scaled by (1-l_full)/(H*E),
// accumulated over 16 e-rows per block -> partial[block][4096]. No atomics.
// grid = 128 = (b,h) x e-quarter.
// ---------------------------------------------------------------------------
__global__ __launch_bounds__(256) void softmax_partial_kernel(
    const float* __restrict__ corr,
    const float* __restrict__ lq1, const float* __restrict__ lk1,
    const float* __restrict__ lq2, const float* __restrict__ lk2,
    float* __restrict__ partial)
{
  __shared__ float accv[4096];
  __shared__ float red[256];
  const int tid = threadIdx.x;
  const int bh  = blockIdx.x >> 2;
  const int eq  = blockIdx.x & 3;

  for (int i = tid; i < 4096; i += 256) accv[i] = 0.f;
  __syncthreads();

  for (int er = 0; er < 16; ++er) {
    const int e = (eq << 4) + er;
    const float* row = corr + (((bh << 6) + e) << 12);
    float v[16], p[16];
    float mx = -3.4e38f;
    #pragma unroll
    for (int k = 0; k < 16; ++k) { v[k] = row[tid + (k << 8)]; mx = fmaxf(mx, v[k]); }
    red[tid] = mx; __syncthreads();
    for (int s = 128; s > 0; s >>= 1) { if (tid < s) red[tid] = fmaxf(red[tid], red[tid + s]); __syncthreads(); }
    mx = red[0]; __syncthreads();
    float sm = 0.f;
    #pragma unroll
    for (int k = 0; k < 16; ++k) { p[k] = __expf(v[k] - mx); sm += p[k]; }
    red[tid] = sm; __syncthreads();
    for (int s = 128; s > 0; s >>= 1) { if (tid < s) red[tid] += red[tid + s]; __syncthreads(); }
    const float inv = 1.0f / red[0];
    __syncthreads();
    #pragma unroll
    for (int k = 0; k < 16; ++k) accv[tid + (k << 8)] += p[k] * inv;  // disjoint slots
  }
  __syncthreads();

  // L_INIT = 0.8 - 0.6*exp(-614.4) == 0.8f in fp32
  const float l1 = __expf(lq1[0] * lk1[0]);
  const float l2 = __expf(lq2[0] * lk2[0]);
  const float sc = (1.f - (l1 - l2 + 0.8f)) * (1.f / 512.f);
  for (int i = tid; i < 4096; i += 256)
    partial[(blockIdx.x << 12) + i] = accv[i] * sc;
}

// ---------------------------------------------------------------------------
// Kernel 3: top-8 of sum_b mean_value, per-b softmax of the gathered weights.
// Single block. Deterministic iterated argmax (lowest-index tie-break).
// ---------------------------------------------------------------------------
__global__ __launch_bounds__(256) void topk_weights_kernel(
    const float* __restrict__ partial, int* __restrict__ idx_out,
    float* __restrict__ w_out)
{
  __shared__ float rv[256];
  __shared__ int   ri[256];
  __shared__ int   sel[8];
  __shared__ float wsh[4][8];
  const int tid = threadIdx.x;

  float mloc[16];
  #pragma unroll
  for (int k = 0; k < 16; ++k) {
    const int tau = tid + (k << 8);
    float s = 0.f;
    for (int jb = 0; jb < 128; ++jb) s += partial[(jb << 12) + tau];
    mloc[k] = s;                       // ranking of sum == ranking of mean (B>0)
  }

  for (int r = 0; r < 8; ++r) {
    float bv = -3.4e38f; int bi = 1 << 30;
    #pragma unroll
    for (int k = 0; k < 16; ++k) {     // ascending tau -> lowest index on ties
      const int tau = tid + (k << 8);
      if (mloc[k] > bv) { bv = mloc[k]; bi = tau; }
    }
    rv[tid] = bv; ri[tid] = bi; __syncthreads();
    for (int s = 128; s > 0; s >>= 1) {
      if (tid < s) {
        if (rv[tid + s] > rv[tid] || (rv[tid + s] == rv[tid] && ri[tid + s] < ri[tid])) {
          rv[tid] = rv[tid + s]; ri[tid] = ri[tid + s];
        }
      }
      __syncthreads();
    }
    const int win = ri[0];
    if (tid == 0) sel[r] = win;
    if ((win & 255) == tid) mloc[win >> 8] = -3.4e38f;
    __syncthreads();
  }

  if (tid < 32) {                      // weights[b,k] = mean_value[b, sel[k]]
    const int b = tid >> 3, k = tid & 7;
    float s = 0.f;
    for (int j = 0; j < 32; ++j) s += partial[(((b << 5) + j) << 12) + sel[k]];
    wsh[b][k] = s;
  }
  __syncthreads();
  if (tid < 4) {                       // softmax over k per b
    float mx = -3.4e38f;
    #pragma unroll
    for (int k = 0; k < 8; ++k) mx = fmaxf(mx, wsh[tid][k]);
    float s = 0.f, ev[8];
    #pragma unroll
    for (int k = 0; k < 8; ++k) { ev[k] = __expf(wsh[tid][k] - mx); s += ev[k]; }
    #pragma unroll
    for (int k = 0; k < 8; ++k) w_out[(tid << 3) + k] = ev[k] / s;
  }
  if (tid < 8) idx_out[tid] = sel[tid];
}

// ---------------------------------------------------------------------------
// Kernel 4: out[b,h,l,e] = sum_k w[b,k] * V[b,h,(l+idx_k)%L, e]  (float4).
// ---------------------------------------------------------------------------
__global__ __launch_bounds__(256) void gather_out_kernel(
    const float* __restrict__ V, const int* __restrict__ idx,
    const float* __restrict__ w, float* __restrict__ out)
{
  const int gid = blockIdx.x * 256 + threadIdx.x;  // float4 index
  const int e4  = gid & 15;
  const int l   = (gid >> 4) & 4095;
  const int bh  = gid >> 16;
  const int b   = bh >> 3;

  int ik[8]; float wk[8];
  #pragma unroll
  for (int k = 0; k < 8; ++k) { ik[k] = idx[k]; wk[k] = w[(b << 3) + k]; }

  const float4* Vb = (const float4*)V + ((size_t)bh << 16);  // bh * L*E/4
  float4 a = make_float4(0.f, 0.f, 0.f, 0.f);
  #pragma unroll
  for (int k = 0; k < 8; ++k) {
    const int row = (l + ik[k]) & 4095;
    const float4 vv = Vb[(row << 4) + e4];
    a.x += wk[k] * vv.x; a.y += wk[k] * vv.y;
    a.z += wk[k] * vv.z; a.w += wk[k] * vv.w;
  }
  ((float4*)out)[gid] = a;
}

// ---------------------------------------------------------------------------
extern "C" void kernel_launch(void* const* d_in, const int* in_sizes, int n_in,
                              void* d_out, int out_size, void* d_ws, size_t ws_size,
                              hipStream_t stream)
{
  (void)in_sizes; (void)n_in; (void)out_size; (void)ws_size;
  const float* Q   = (const float*)d_in[0];
  const float* K   = (const float*)d_in[1];
  const float* V   = (const float*)d_in[2];
  const float* lq1 = (const float*)d_in[3];
  const float* lk1 = (const float*)d_in[4];
  const float* lq2 = (const float*)d_in[5];
  const float* lk2 = (const float*)d_in[6];
  float* out = (float*)d_out;

  char* ws = (char*)d_ws;                      // ~36 MB used
  float* corr    = (float*)ws;                             // 33,554,432 B
  float* partial = (float*)(ws + 33554432);                //  2,097,152 B
  int*   idxb    = (int*)  (ws + 33554432 + 2097152);      //         32 B
  float* wts     = (float*)(ws + 33554432 + 2097152 + 256);//        128 B

  corr_wmma_kernel      <<<2048, 512, 0, stream>>>(Q, K, corr);
  softmax_partial_kernel<<< 128, 256, 0, stream>>>(corr, lq1, lk1, lq2, lk2, partial);
  topk_weights_kernel   <<<   1, 256, 0, stream>>>(partial, idxb, wts);
  gather_out_kernel     <<<8192, 256, 0, stream>>>(V, idxb, wts, out);
}
